// GATEncoder_5806795784251
// MI455X (gfx1250) — matrix operand order repaired and checked by reading, not compiled, not run on hardware
//
#include <hip/hip_runtime.h>
#include <math.h>

#define NEGS 0.2f
#define BN_EPS 1e-5f

typedef __attribute__((ext_vector_type(16))) __bf16 v16bf;
typedef __attribute__((ext_vector_type(8)))  float  v8f;

// ---------------------------------------------------------------------------
// Small helpers
// ---------------------------------------------------------------------------
__device__ __forceinline__ float leaky(float v) { return v > 0.f ? v : v * NEGS; }

__device__ __forceinline__ void atomicMaxF(float* addr, float v) {
    // int/uint ordering trick; addr initialized to -inf
    if (v >= 0.f) atomicMax((int*)addr, __float_as_int(v));
    else          atomicMin((unsigned int*)addr, __float_as_uint(v));
}

// ---------------------------------------------------------------------------
// Elementwise conversion kernels
// ---------------------------------------------------------------------------
__global__ void k_f32_to_bf16(const float* __restrict__ in, __bf16* __restrict__ out, int n) {
    int i = blockIdx.x * blockDim.x + threadIdx.x;
    if (i < n) out[i] = (__bf16)in[i];
}

// W[Fin][Fout] fp32 -> Wt[Fout][Fin] bf16
__global__ void k_wt_transpose(const float* __restrict__ W, __bf16* __restrict__ Wt,
                               int Fin, int Fout) {
    int i = blockIdx.x * blockDim.x + threadIdx.x;
    if (i >= Fin * Fout) return;
    int r = i / Fout, c = i % Fout;
    Wt[(size_t)c * Fin + r] = (__bf16)W[i];
}

__global__ void k_fill_f32(float* __restrict__ p, float v, int n) {
    int i = blockIdx.x * blockDim.x + threadIdx.x;
    if (i < n) p[i] = v;
}

// ---------------------------------------------------------------------------
// WMMA bf16 GEMM:  out[N,Fout] = A[N,Fin] (bf16) x Bt[Fout,Fin]^T (bf16)
// Wave computes a 32-row x 64-col strip = 2x4 grid of 16x16 WMMA tiles.
// Per k-step: 2 A frags + 4 B frags -> 8 WMMAs (6 dwords loaded per WMMA).
// MODE 0: f32 out, no bias | MODE 1: bias+leaky, bf16 out | MODE 2: bias, f32 out
// ---------------------------------------------------------------------------
template <int MODE>
__global__ __launch_bounds__(256)
void k_wmma_gemm(const __bf16* __restrict__ A, const __bf16* __restrict__ Bt,
                 const float* __restrict__ bias, void* __restrict__ out,
                 int Nrows, int Fin, int Fout) {
    const int lane = threadIdx.x & 31;
    const int wave = blockIdx.x * (blockDim.x >> 5) + (threadIdx.x >> 5);
    const int rtiles  = Nrows >> 4;
    const int rgroups = (rtiles + 1) >> 1;   // 2 row tiles per wave (last may be single)
    const int cgroups = Fout >> 6;           // 4 col tiles per wave (Fout % 64 == 0)
    if (wave >= rgroups * cgroups) return;   // wave-uniform: EXEC all-ones below
    const int rg = wave / cgroups, cg = wave % cgroups;
    const int half = lane >> 4;
    const int l16  = lane & 15;

    const bool t1ok = (rg * 2 + 1) < rtiles;         // second row tile valid?
    int row0 = rg * 32 + l16;
    int row1 = row0 + 16;
    if (row1 >= Nrows) row1 = Nrows - 1;             // clamp loads; stores guarded
    const size_t arow0 = (size_t)row0 * Fin;
    const size_t arow1 = (size_t)row1 * Fin;
    size_t brow[4];
#pragma unroll
    for (int t = 0; t < 4; ++t) brow[t] = (size_t)(cg * 64 + t * 16 + l16) * Fin;

    v8f acc[2][4] = {};
    for (int k0 = 0; k0 < Fin; k0 += 32) {
        union U { unsigned int u[8]; v16bf v; };
        U a0, a1, b[4];
#pragma unroll
        for (int j = 0; j < 8; ++j) {
            // A 16x32 frag, contiguous k order chosen here: K = 16*half + 2*j
            const int ka = k0 + half * 16 + 2 * j;
            // B 32x16 frag, same contiguous k order as A: K = 16*half + 2*j
            const int kb = k0 + half * 16 + 2 * j;
            a0.u[j] = *(const unsigned int*)(A + arow0 + ka);
            a1.u[j] = *(const unsigned int*)(A + arow1 + ka);
#pragma unroll
            for (int t = 0; t < 4; ++t)
                b[t].u[j] = *(const unsigned int*)(Bt + brow[t] + kb);
        }
#pragma unroll
        for (int t = 0; t < 4; ++t) {
            acc[0][t] = __builtin_amdgcn_wmma_f32_16x16x32_bf16(
                false, a0.v, false, b[t].v, (short)0, acc[0][t], false, false);
            acc[1][t] = __builtin_amdgcn_wmma_f32_16x16x32_bf16(
                false, a1.v, false, b[t].v, (short)0, acc[1][t], false, false);
        }
    }

#pragma unroll
    for (int i = 0; i < 2; ++i) {
        if (i == 1 && !t1ok) break;              // wave-uniform guard
#pragma unroll
        for (int t = 0; t < 4; ++t) {
            const int n = cg * 64 + t * 16 + l16;
            const float bv = (MODE != 0) ? bias[n] : 0.f;
#pragma unroll
            for (int r = 0; r < 8; ++r) {
                const int m = rg * 32 + i * 16 + r + half * 8;  // C/D: VGPR r -> M=r+8*half
                float v = acc[i][t][r] + bv;
                if (MODE == 1) v = leaky(v);
                if (MODE == 1) ((__bf16*)out)[(size_t)m * Fout + n] = (__bf16)v;
                else           ((float*)out)[(size_t)m * Fout + n] = v;
            }
        }
    }
}

// ---------------------------------------------------------------------------
// Attention scores: wave per (n,h); lanes stride channels (coalesced), then
// shuffle-reduce. a_s[n,h] = sum_c h[n,h,c]*att_s[h,c], same for a_d.
// ---------------------------------------------------------------------------
__global__ void k_att_scores(const float* __restrict__ h, const float* __restrict__ att_s,
                             const float* __restrict__ att_d, float* __restrict__ as_o,
                             float* __restrict__ ad_o, int Nn, int H, int C) {
    int lane = threadIdx.x & 31;
    int w = blockIdx.x * (blockDim.x >> 5) + (threadIdx.x >> 5);
    if (w >= Nn * H) return;
    int hh = w % H;
    const float* row = h + (size_t)w * C;
    const float* wsp = att_s + hh * C;
    const float* wdp = att_d + hh * C;
    float ss = 0.f, sd = 0.f;
    for (int c = lane; c < C; c += 32) {
        float v = row[c]; ss += v * wsp[c]; sd += v * wdp[c];
    }
#pragma unroll
    for (int off = 16; off > 0; off >>= 1) {
        ss += __shfl_xor(ss, off, 32);
        sd += __shfl_xor(sd, off, 32);
    }
    if (lane == 0) { as_o[w] = ss; ad_o[w] = sd; }
}

__device__ __forceinline__ int esrc(const int* ei, int E, int i) { return i < E ? ei[i]     : i - E; }
__device__ __forceinline__ int edst(const int* ei, int E, int i) { return i < E ? ei[E + i] : i - E; }

__global__ void k_edge_max(const int* __restrict__ ei, int E, int Etot,
                           const float* __restrict__ as_, const float* __restrict__ ad_,
                           float* __restrict__ m, int H) {
    int e = blockIdx.x * blockDim.x + threadIdx.x;
    if (e >= Etot) return;
    int s = esrc(ei, E, e), d = edst(ei, E, e);
    for (int h = 0; h < H; ++h)
        atomicMaxF(&m[d * H + h], leaky(as_[s * H + h] + ad_[d * H + h]));
}

__global__ void k_edge_exp(const int* __restrict__ ei, int E, int Etot,
                           const float* __restrict__ as_, const float* __restrict__ ad_,
                           const float* __restrict__ m, float* __restrict__ ez,
                           float* __restrict__ den, int H) {
    int e = blockIdx.x * blockDim.x + threadIdx.x;
    if (e >= Etot) return;
    int s = esrc(ei, E, e), d = edst(ei, E, e);
    for (int h = 0; h < H; ++h) {
        float v = leaky(as_[s * H + h] + ad_[d * H + h]);
        float z = expf(v - m[d * H + h]);
        ez[(size_t)e * H + h] = z;
        atomicAdd(&den[d * H + h], z);
    }
}

// wave per edge; lanes stride the H*C feature channels (coalesced atomics)
__global__ void k_edge_scatter(const int* __restrict__ ei, int E, int Etot,
                               const float* __restrict__ hsrc, const float* __restrict__ ez,
                               const float* __restrict__ den, float* __restrict__ out,
                               int H, int C) {
    int lane = threadIdx.x & 31;
    int e = blockIdx.x * (blockDim.x >> 5) + (threadIdx.x >> 5);
    if (e >= Etot) return;
    int s = esrc(ei, E, e), d = edst(ei, E, e);
    const int F = H * C;
    const float* hs = hsrc + (size_t)s * F;
    float* od = out + (size_t)d * F;
    for (int f = lane; f < F; f += 32) {
        int h = f / C;
        float alpha = ez[(size_t)e * H + h] / (den[d * H + h] + 1e-16f);
        atomicAdd(&od[f], hs[f] * alpha);
    }
}

// x1_bf = bf16(agg + bias)   (concat layer output feeding next GEMM)
__global__ void k_bias_bf16(const float* __restrict__ agg, const float* __restrict__ b,
                            __bf16* __restrict__ out, int Nn, int F) {
    int i = blockIdx.x * blockDim.x + threadIdx.x;
    if (i >= Nn * F) return;
    out[i] = (__bf16)(agg[i] + b[i % F]);
}

// x2_bf[n,c] = bf16(mean_h agg[n,h,c] + b[c])
__global__ void k_headmean_bf16(const float* __restrict__ agg, const float* __restrict__ b,
                                __bf16* __restrict__ out, int Nn, int H, int C) {
    int i = blockIdx.x * blockDim.x + threadIdx.x;
    if (i >= Nn * C) return;
    int n = i / C, c = i % C;
    float acc = 0.f;
    for (int h = 0; h < H; ++h) acc += agg[((size_t)n * H + h) * C + c];
    out[i] = (__bf16)(acc / (float)H + b[c]);
}

// ---------------------------------------------------------------------------
// BatchNorm: one block per column -> stats[c]=mu, stats[C+c]=biased var
// ---------------------------------------------------------------------------
__global__ void k_bn_stats(const float* __restrict__ y, float* __restrict__ stats,
                           int Nn, int Cc) {
    __shared__ float ssum[256], ssq[256];
    int c = blockIdx.x, t = threadIdx.x;
    float s = 0.f, q = 0.f;
    for (int n = t; n < Nn; n += blockDim.x) {
        float v = y[(size_t)n * Cc + c]; s += v; q += v * v;
    }
    ssum[t] = s; ssq[t] = q;
    __syncthreads();
    for (int st = 128; st > 0; st >>= 1) {
        if (t < st) { ssum[t] += ssum[t + st]; ssq[t] += ssq[t + st]; }
        __syncthreads();
    }
    if (t == 0) {
        float mu = ssum[0] / (float)Nn;
        stats[c] = mu;
        stats[Cc + c] = ssq[0] / (float)Nn - mu * mu;
    }
}

__global__ void k_bn_apply(const float* __restrict__ y, const float* __restrict__ stats,
                           const float* __restrict__ gamma, const float* __restrict__ beta,
                           float* __restrict__ out, int Nn, int Cc) {
    int i = blockIdx.x * blockDim.x + threadIdx.x;
    if (i >= Nn * Cc) return;
    int c = i % Cc;
    out[i] = (y[i] - stats[c]) * rsqrtf(stats[Cc + c] + BN_EPS) * gamma[c] + beta[c];
}

// ---------------------------------------------------------------------------
// Host orchestration
// ---------------------------------------------------------------------------
static inline int ceil_div(long long a, int b) { return (int)((a + b - 1) / b); }

template <int MODE>
static void launch_gemm(const __bf16* A, const __bf16* Bt, const float* bias, void* out,
                        int Nrows, int Fin, int Fout, hipStream_t s) {
    int rtiles  = Nrows / 16;
    int rgroups = (rtiles + 1) / 2;
    int cgroups = Fout / 64;
    int waves   = rgroups * cgroups;
    k_wmma_gemm<MODE><<<ceil_div(waves, 8), 256, 0, s>>>(A, Bt, bias, out, Nrows, Fin, Fout);
}

extern "C" void kernel_launch(void* const* d_in, const int* in_sizes, int n_in,
                              void* d_out, int out_size, void* d_ws, size_t ws_size,
                              hipStream_t stream) {
    const float* attr = (const float*)d_in[0];
    const int*   ei   = (const int*)d_in[1];
    const float* Wprep= (const float*)d_in[2];  const float* bprep = (const float*)d_in[3];
    const float* W1   = (const float*)d_in[4];  const float* b1    = (const float*)d_in[5];
    const float* as1  = (const float*)d_in[6];  const float* ad1   = (const float*)d_in[7];
    const float* W2   = (const float*)d_in[8];  const float* b2    = (const float*)d_in[9];
    const float* as2  = (const float*)d_in[10]; const float* ad2   = (const float*)d_in[11];
    const float* Wp1  = (const float*)d_in[12]; const float* bp1   = (const float*)d_in[13];
    const float* Wp2  = (const float*)d_in[14]; const float* bp2   = (const float*)d_in[15];
    const float* gamma= (const float*)d_in[16]; const float* beta  = (const float*)d_in[17];

    const int N = in_sizes[0] / 64;      // 50000
    const int E = in_sizes[1] / 2;       // 800000
    const int Etot = E + N;

    // ---- workspace layout (aliased across phases) ----
    char* ws = (char*)d_ws;
    size_t o = 0;
    auto alloc = [&](size_t bytes) { size_t r = o; o += (bytes + 255) & ~(size_t)255; return r; };
    size_t o_h    = alloc((size_t)N * 512 * 4);   // h1 (f32), later h2
    size_t o_agg  = alloc((size_t)N * 512 * 4);   // out1 agg, later out2 agg
    size_t o_x1   = alloc((size_t)N * 512 * 2);   // x1 bf16
    size_t o_x256 = alloc((size_t)N * 256 * 2);   // x0 bf16, later y1 bf16
    size_t o_ez   = alloc((size_t)Etot * 4 * 4);  // edge exp
    size_t o_x64  = alloc((size_t)N * 64 * 2);    // attr bf16, later x2 bf16
    size_t o_as   = alloc((size_t)N * 4 * 4);
    size_t o_ad   = alloc((size_t)N * 4 * 4);
    size_t o_m    = alloc((size_t)N * 4 * 4);
    size_t o_den  = alloc((size_t)N * 4 * 4);
    size_t o_y2   = alloc((size_t)N * 64 * 4);    // pre-BN output f32
    size_t o_wtp  = alloc(64 * 256 * 2);
    size_t o_wt1  = alloc(256 * 512 * 2);
    size_t o_wt2  = alloc(512 * 256 * 2);
    size_t o_wq1  = alloc(64 * 256 * 2);
    size_t o_wq2  = alloc(256 * 64 * 2);
    size_t o_stat = alloc(128 * 4);
    (void)ws_size;

    __bf16* attr_bf = (__bf16*)(ws + o_x64);
    __bf16* x256_bf = (__bf16*)(ws + o_x256);
    __bf16* x1_bf   = (__bf16*)(ws + o_x1);
    float*  hbuf    = (float*)(ws + o_h);
    float*  agg     = (float*)(ws + o_agg);
    float*  ezb     = (float*)(ws + o_ez);
    float*  asb     = (float*)(ws + o_as);
    float*  adb     = (float*)(ws + o_ad);
    float*  mb      = (float*)(ws + o_m);
    float*  denb    = (float*)(ws + o_den);
    float*  y2      = (float*)(ws + o_y2);
    float*  stats   = (float*)(ws + o_stat);

    const int T = 256;
    // ---- weight prep (bf16, transposed) + attribute conversion ----
    k_f32_to_bf16<<<ceil_div((long long)N * 64, T), T, 0, stream>>>(attr, attr_bf, N * 64);
    k_wt_transpose<<<ceil_div(64 * 256, T), T, 0, stream>>>(Wprep, (__bf16*)(ws + o_wtp), 64, 256);
    k_wt_transpose<<<ceil_div(256 * 512, T), T, 0, stream>>>(W1, (__bf16*)(ws + o_wt1), 256, 512);
    k_wt_transpose<<<ceil_div(512 * 256, T), T, 0, stream>>>(W2, (__bf16*)(ws + o_wt2), 512, 256);
    k_wt_transpose<<<ceil_div(64 * 256, T), T, 0, stream>>>(Wp1, (__bf16*)(ws + o_wq1), 64, 256);
    k_wt_transpose<<<ceil_div(256 * 64, T), T, 0, stream>>>(Wp2, (__bf16*)(ws + o_wq2), 256, 64);

    // ---- prep MLP: x0 = leaky(attr @ Wprep + bprep) -> bf16 ----
    launch_gemm<1>(attr_bf, (__bf16*)(ws + o_wtp), bprep, x256_bf, N, 64, 256, stream);

    // ---- GAT layer 1 ----
    launch_gemm<0>(x256_bf, (__bf16*)(ws + o_wt1), nullptr, hbuf, N, 256, 512, stream);
    k_att_scores<<<ceil_div((long long)N * 4, 8), T, 0, stream>>>(hbuf, as1, ad1, asb, adb, N, 4, 128);
    k_fill_f32<<<ceil_div((long long)N * 4, T), T, 0, stream>>>(mb, -INFINITY, N * 4);
    k_fill_f32<<<ceil_div((long long)N * 4, T), T, 0, stream>>>(denb, 0.f, N * 4);
    k_fill_f32<<<ceil_div((long long)N * 512, T), T, 0, stream>>>(agg, 0.f, N * 512);
    k_edge_max<<<ceil_div(Etot, T), T, 0, stream>>>(ei, E, Etot, asb, adb, mb, 4);
    k_edge_exp<<<ceil_div(Etot, T), T, 0, stream>>>(ei, E, Etot, asb, adb, mb, ezb, denb, 4);
    k_edge_scatter<<<ceil_div(Etot, 8), T, 0, stream>>>(ei, E, Etot, hbuf, ezb, denb, agg, 4, 128);
    k_bias_bf16<<<ceil_div((long long)N * 512, T), T, 0, stream>>>(agg, b1, x1_bf, N, 512);

    // ---- GAT layer 2 ----
    launch_gemm<0>(x1_bf, (__bf16*)(ws + o_wt2), nullptr, hbuf, N, 512, 256, stream);
    k_att_scores<<<ceil_div((long long)N * 4, 8), T, 0, stream>>>(hbuf, as2, ad2, asb, adb, N, 4, 64);
    k_fill_f32<<<ceil_div((long long)N * 4, T), T, 0, stream>>>(mb, -INFINITY, N * 4);
    k_fill_f32<<<ceil_div((long long)N * 4, T), T, 0, stream>>>(denb, 0.f, N * 4);
    k_fill_f32<<<ceil_div((long long)N * 256, T), T, 0, stream>>>(agg, 0.f, N * 256);
    k_edge_max<<<ceil_div(Etot, T), T, 0, stream>>>(ei, E, Etot, asb, adb, mb, 4);
    k_edge_exp<<<ceil_div(Etot, T), T, 0, stream>>>(ei, E, Etot, asb, adb, mb, ezb, denb, 4);
    k_edge_scatter<<<ceil_div(Etot, 8), T, 0, stream>>>(ei, E, Etot, hbuf, ezb, denb, agg, 4, 64);
    k_headmean_bf16<<<ceil_div((long long)N * 64, T), T, 0, stream>>>(agg, b2, (__bf16*)(ws + o_x64), N, 4, 64);

    // ---- post MLP ----
    launch_gemm<1>((__bf16*)(ws + o_x64), (__bf16*)(ws + o_wq1), bp1, x256_bf, N, 64, 256, stream);
    launch_gemm<2>(x256_bf, (__bf16*)(ws + o_wq2), bp2, y2, N, 256, 64, stream);

    // ---- BatchNorm ----
    k_bn_stats<<<64, 256, 0, stream>>>(y2, stats, N, 64);
    k_bn_apply<<<ceil_div((long long)N * 64, T), T, 0, stream>>>(y2, stats, gamma, beta,
                                                                 (float*)d_out, N, 64);
}
